// WindowAttention_32830730010663
// MI455X (gfx1250) — matrix operand order repaired and checked by reading, not compiled, not run on hardware
//
#include <hip/hip_runtime.h>
#include <hip/hip_bf16.h>

// ---------------------------------------------------------------------------
// Fused window-attention (CSWin/LePE style) for MI455X (gfx1250, wave32).
// bf16 WMMA 16x16x32 for all GEMMs, f32 accumulate. One block per window.
// Dynamic LDS = 168 KB (CDNA5 WGP has 320 KB).
// ---------------------------------------------------------------------------

typedef __attribute__((ext_vector_type(16))) __bf16 v16bf;
typedef __attribute__((ext_vector_type(8)))  float  v8f;

#define NTOK 64
#define CCH  192
#define NHD  6
#define HDD  32
#define QSCALE 0.17677669529663689f   // 1/sqrt(32)

// LDS byte offsets (all 16B aligned)
#define OFF_Q   0        // q bf16 [64][192]            24576  (later dwp f32 low half)
#define OFF_K   24576    // k bf16 [64][192]            24576  (later dwp f32 high half)
#define OFF_VT  49152    // v^T bf16 [192][64]          24576
#define OFF_VR  73728    // v_r f32 [192][64]           49152
#define OFF_XB  122880   // x bf16 [64][192], later O/Afin bf16
#define OFF_S   147456   // scores f32 [64][64]         16384 (per head)
#define OFF_P   163840   // probs bf16 [64][64]          8192
#define SMEM_BYTES 172032

union ABu { uint4 u[2]; v16bf v; };

__device__ __forceinline__ v8f wmma_bf16(const ABu& a, const ABu& b, v8f c) {
  // 8 args: neg_a, A, neg_b, B, c_mod, C, reuse_a, reuse_b
  return __builtin_amdgcn_wmma_f32_16x16x32_bf16(false, a.v, false, b.v,
                                                 (short)0, c, false, false);
}

// Pack [wq | wkv | proj_w] -> Wt[out(960)][in(192)] bf16 (B-operand friendly:
// per output column, K is contiguous).
__global__ void pack_weights(const float* __restrict__ wq,
                             const float* __restrict__ wkv,
                             const float* __restrict__ prw,
                             __bf16* __restrict__ wt) {
  int out = blockIdx.x;      // 0..959
  int in  = threadIdx.x;     // 0..191
  float v;
  if (out < 192)      v = wq [in * 192 + out];
  else if (out < 768) v = wkv[in * 576 + (out - 192)];
  else                v = prw[in * 192 + (out - 768)];
  wt[(size_t)out * 192 + in] = (__bf16)v;
}

__global__ void __launch_bounds__(256)
win_attn_fused(const float* __restrict__ x, const __bf16* __restrict__ wt,
               const float* __restrict__ bq, const float* __restrict__ bkv,
               const float* __restrict__ rpb, const int* __restrict__ relidx,
               const float* __restrict__ dww, const float* __restrict__ dwb,
               const float* __restrict__ pww, const float* __restrict__ pwb,
               const float* __restrict__ projb, float* __restrict__ out) {
  extern __shared__ char smem[];
  __bf16* qs  = (__bf16*)(smem + OFF_Q);
  __bf16* ks  = (__bf16*)(smem + OFF_K);
  __bf16* vt  = (__bf16*)(smem + OFF_VT);
  float*  vr  = (float*) (smem + OFF_VR);
  __bf16* xb  = (__bf16*)(smem + OFF_XB);   // x, then O / (O+position)
  float*  Sf  = (float*) (smem + OFF_S);
  __bf16* Pb  = (__bf16*)(smem + OFF_P);
  float*  dwp = (float*) (smem + OFF_Q);    // aliases q+k after attention

  const int tid = threadIdx.x;
  const int wv  = tid >> 5;
  const int ln  = tid & 31;
  const int lh  = ln >> 4;      // lane[4]: K-half / M-half selector
  const int l16 = ln & 15;      // row (A) / column (B,C,D)
  const int b   = blockIdx.x;
  const float* xg = x + (size_t)b * (NTOK * CCH);

  // ---- Phase 1: window of x -> LDS as bf16 --------------------------------
  for (int i = 0; i < 12; ++i) {
    float4 f = ((const float4*)xg)[tid + i * 256];
    int e = (tid + i * 256) * 4;
    xb[e + 0] = (__bf16)f.x; xb[e + 1] = (__bf16)f.y;
    xb[e + 2] = (__bf16)f.z; xb[e + 3] = (__bf16)f.w;
  }
  __syncthreads();

  // ---- Phase 2: QKV+Vr GEMM  (64x192) @ (192x768) -------------------------
  for (int job = wv; job < 192; job += 8) {
    int nt = job >> 2, mt = job & 3;
    int ncol = nt * 16 + l16;
    int arow = mt * 16 + l16;
    const __bf16* wrow = wt + (size_t)ncol * CCH;
    v8f acc = {0.f, 0.f, 0.f, 0.f, 0.f, 0.f, 0.f, 0.f};
    for (int k0 = 0; k0 < CCH; k0 += 32) {
      ABu a, bb;
      a.u[0]  = *(const uint4*)(xb + arow * CCH + k0 + lh * 8);
      a.u[1]  = *(const uint4*)(xb + arow * CCH + k0 + 16 + lh * 8);
      bb.u[0] = *(const uint4*)(wrow + k0 + lh * 8);
      bb.u[1] = *(const uint4*)(wrow + k0 + 16 + lh * 8);
      acc = wmma_bf16(a, bb, acc);
    }
    float cb = (ncol < CCH) ? bq[ncol] : bkv[ncol - CCH];
    if (ncol < 192) {                         // q (pre-scaled)
      for (int r = 0; r < 8; ++r) {
        int tok = mt * 16 + lh * 8 + r;
        qs[tok * CCH + ncol] = (__bf16)((acc[r] + cb) * QSCALE);
      }
    } else if (ncol < 384) {                  // k, row-major
      int c = ncol - 192;
      for (int r = 0; r < 8; ++r) {
        int tok = mt * 16 + lh * 8 + r;
        ks[tok * CCH + c] = (__bf16)(acc[r] + cb);
      }
    } else if (ncol < 576) {                  // v, transposed [c][tok]
      int c = ncol - 384;
      __bf16* dst = vt + c * NTOK + mt * 16 + lh * 8;
      for (int r = 0; r < 8; ++r) dst[r] = (__bf16)(acc[r] + cb);
    } else {                                  // v_r, channel-major f32
      int c = ncol - 576;
      float* dst = vr + c * NTOK + mt * 16 + lh * 8;
      for (int r = 0; r < 8; ++r) dst[r] = acc[r] + cb;
    }
  }
  __syncthreads();

  // ---- Phase 3: per-head attention (hd=32 -> one WMMA K-step) -------------
  for (int h = 0; h < NHD; ++h) {
    const int k0h = h * HDD;
    // 3a: S = q_h @ k_h^T + rel_bias
    for (int job = wv; job < 16; job += 8) {
      int mt = job >> 2, nt = job & 3;
      int arow = mt * 16 + l16, brow = nt * 16 + l16;
      ABu a, bb;
      a.u[0]  = *(const uint4*)(qs + arow * CCH + k0h + lh * 8);
      a.u[1]  = *(const uint4*)(qs + arow * CCH + k0h + 16 + lh * 8);
      bb.u[0] = *(const uint4*)(ks + brow * CCH + k0h + lh * 8);
      bb.u[1] = *(const uint4*)(ks + brow * CCH + k0h + 16 + lh * 8);
      v8f acc = {0.f, 0.f, 0.f, 0.f, 0.f, 0.f, 0.f, 0.f};
      acc = wmma_bf16(a, bb, acc);
      int ncol = nt * 16 + l16;
      for (int r = 0; r < 8; ++r) {
        int tok = mt * 16 + lh * 8 + r;
        int idx = relidx[tok * NTOK + ncol];
        Sf[tok * NTOK + ncol] = acc[r] + rpb[idx * NHD + h];
      }
    }
    __syncthreads();
    // 3b: row softmax -> P (bf16)
    if (tid < NTOK) {
      float m = -1e30f;
      for (int j = 0; j < NTOK; ++j) m = fmaxf(m, Sf[tid * NTOK + j]);
      float s = 0.f;
      for (int j = 0; j < NTOK; ++j) {
        float e = __expf(Sf[tid * NTOK + j] - m);
        s += e;
        Sf[tid * NTOK + j] = e;
      }
      float inv = 1.f / s;
      for (int j = 0; j < NTOK; ++j)
        Pb[tid * NTOK + j] = (__bf16)(Sf[tid * NTOK + j] * inv);
    }
    __syncthreads();
    // 3c: O_h = P @ v_h  (4 Mtiles x 2 Ntiles = 8 jobs = 8 waves)
    {
      int mt = wv >> 1, ntd = wv & 1;
      int arow = mt * 16 + l16;
      int c = k0h + ntd * 16 + l16;
      v8f acc = {0.f, 0.f, 0.f, 0.f, 0.f, 0.f, 0.f, 0.f};
      for (int k0 = 0; k0 < NTOK; k0 += 32) {
        ABu a, bb;
        a.u[0]  = *(const uint4*)(Pb + arow * NTOK + k0 + lh * 8);
        a.u[1]  = *(const uint4*)(Pb + arow * NTOK + k0 + 16 + lh * 8);
        bb.u[0] = *(const uint4*)(vt + c * NTOK + k0 + lh * 8);
        bb.u[1] = *(const uint4*)(vt + c * NTOK + k0 + 16 + lh * 8);
        acc = wmma_bf16(a, bb, acc);
      }
      for (int r = 0; r < 8; ++r) {
        int tok = mt * 16 + lh * 8 + r;
        xb[tok * CCH + c] = (__bf16)acc[r];   // xb region now holds O
      }
    }
    __syncthreads();
  }

  // ---- Phase 4a: depthwise 5x5 'SAME' on vr (8x8 grid) --------------------
  for (int i = 0; i < 48; ++i) {
    int idx = tid + i * 256;
    int c = idx >> 6, pix = idx & 63;
    int py = pix >> 3, px = pix & 7;
    float sum = dwb[c];
    const float* w5 = dww + c * 25;
    for (int ky = 0; ky < 5; ++ky) {
      int yy = py + ky - 2;
      if (yy < 0 || yy > 7) continue;
      for (int kx = 0; kx < 5; ++kx) {
        int xx = px + kx - 2;
        if (xx < 0 || xx > 7) continue;
        sum += vr[c * 64 + yy * 8 + xx] * w5[ky * 5 + kx];
      }
    }
    dwp[c * 64 + pix] = sum;
  }
  __syncthreads();
  // ---- Phase 4b: grouped 1x1 (groups=6) + add into O ----------------------
  for (int i = 0; i < 48; ++i) {
    int idx = tid + i * 256;
    int c = idx >> 6, pix = idx & 63;
    int g = c >> 5;
    float sum = pwb[c];
    const float* pwr = pww + c * 32;
    const float* dp = dwp + g * 32 * 64 + pix;
    for (int j = 0; j < 32; ++j) sum += dp[j * 64] * pwr[j];
    float o = (float)xb[pix * CCH + c] + sum;
    xb[pix * CCH + c] = (__bf16)o;            // Afin = O + position
  }
  __syncthreads();

  // ---- Phase 5: out = Afin @ proj_w + proj_b ------------------------------
  float* og = out + (size_t)b * (NTOK * CCH);
  for (int job = wv; job < 48; job += 8) {
    int nt = job >> 2, mt = job & 3;
    int ncol = nt * 16 + l16;
    int arow = mt * 16 + l16;
    const __bf16* wrow = wt + (size_t)(768 + ncol) * CCH;
    v8f acc = {0.f, 0.f, 0.f, 0.f, 0.f, 0.f, 0.f, 0.f};
    for (int k0 = 0; k0 < CCH; k0 += 32) {
      ABu a, bb;
      a.u[0]  = *(const uint4*)(xb + arow * CCH + k0 + lh * 8);
      a.u[1]  = *(const uint4*)(xb + arow * CCH + k0 + 16 + lh * 8);
      bb.u[0] = *(const uint4*)(wrow + k0 + lh * 8);
      bb.u[1] = *(const uint4*)(wrow + k0 + 16 + lh * 8);
      acc = wmma_bf16(a, bb, acc);
    }
    float cb = projb[ncol];
    for (int r = 0; r < 8; ++r) {
      int tok = mt * 16 + lh * 8 + r;
      og[tok * CCH + ncol] = acc[r] + cb;
    }
  }
}

extern "C" void kernel_launch(void* const* d_in, const int* in_sizes, int n_in,
                              void* d_out, int out_size, void* d_ws, size_t ws_size,
                              hipStream_t stream) {
  const float* x    = (const float*)d_in[0];
  const float* wq   = (const float*)d_in[1];
  const float* bq   = (const float*)d_in[2];
  const float* wkv  = (const float*)d_in[3];
  const float* bkv  = (const float*)d_in[4];
  const float* rpb  = (const float*)d_in[5];
  const int*   rel  = (const int*)d_in[6];
  const float* dww  = (const float*)d_in[7];
  const float* dwb  = (const float*)d_in[8];
  const float* pww  = (const float*)d_in[9];
  const float* pwb  = (const float*)d_in[10];
  const float* prw  = (const float*)d_in[11];
  const float* prb  = (const float*)d_in[12];
  float*   outp = (float*)d_out;
  __bf16*  wtp  = (__bf16*)d_ws;   // 960*192*2 = 368,640 bytes of workspace

  pack_weights<<<960, 192, 0, stream>>>(wq, wkv, prw, wtp);
  win_attn_fused<<<4096, 256, SMEM_BYTES, stream>>>(
      x, wtp, bq, bkv, rpb, rel, dww, dwb, pww, pwb, prb, outp);
}
